// Conv2d_int8_STE_58128087384363
// MI455X (gfx1250) — 8-bit matrix operand order repaired and checked by reading, compiled, not run on hardware
//
#include <hip/hip_runtime.h>
#include <hip/hip_bf16.h>


typedef __attribute__((ext_vector_type(8))) int v8i;

#define K_PAD   320
#define M_TOT   4096
#define N_TOT   64
#define K_REAL  288

__device__ __forceinline__ int quant8(float v, float inv_scale) {
    float r = rintf(v * inv_scale);
    r = fminf(fmaxf(r, -128.f), 127.f);
    return (int)r;
}

__global__ void k_zero(float* s) {
    if (threadIdx.x < 2) s[threadIdx.x] = 0.f;
}

__global__ void k_absmax(const float* __restrict__ src, int n,
                         unsigned* __restrict__ dst) {
    __shared__ float smem[8];
    float v = 0.f;
    for (int i = blockIdx.x * blockDim.x + threadIdx.x; i < n;
         i += gridDim.x * blockDim.x)
        v = fmaxf(v, fabsf(src[i]));
    #pragma unroll
    for (int off = 16; off; off >>= 1)
        v = fmaxf(v, __shfl_xor(v, off, 32));
    int lane = threadIdx.x & 31, wv = threadIdx.x >> 5;
    if (lane == 0) smem[wv] = v;
    __syncthreads();
    if (wv == 0) {
        v = (lane < (int)(blockDim.x >> 5)) ? smem[lane] : 0.f;
        #pragma unroll
        for (int off = 4; off; off >>= 1)
            v = fmaxf(v, __shfl_xor(v, off, 32));
        if (lane == 0) atomicMax(dst, __float_as_uint(v));
    }
}

__global__ void k_pack_a(const float* __restrict__ x,
                         const float* __restrict__ absmax,
                         signed char* __restrict__ A) {
    int idx = blockIdx.x * 256 + threadIdx.x;
    int k = idx % K_PAD;
    int m = idx / K_PAD;
    float am = absmax[0];
    float inv = (am > 0.f) ? (127.f / am) : 0.f;
    signed char q = 0;
    if (k < K_REAL) {
        int c  = k / 9, r = k - c * 9;
        int kh = r / 3, kw = r - kh * 3;
        int b  = m >> 10, pix = m & 1023;
        int oh = pix >> 5, ow = pix & 31;
        int ih = oh + kh - 1, iw = ow + kw - 1;
        if ((unsigned)ih < 32u && (unsigned)iw < 32u)
            q = (signed char)quant8(x[((b * 32 + c) * 32 + ih) * 32 + iw], inv);
    }
    A[idx] = q;
}

__global__ void k_pack_b(const float* __restrict__ w,
                         const float* __restrict__ absmax,
                         signed char* __restrict__ Bp) {
    int idx = blockIdx.x * 256 + threadIdx.x;
    int byteInLane = idx & 31;
    int v    = byteInLane >> 2;
    int j    = byteInLane & 3;
    int lane = (idx >> 5) & 31;
    int t    = idx >> 10;
    int kt   = t % 5;
    int nt   = t / 5;
    int k = kt * 64 + ((v >> 1) * 16 + (v & 1) * 4 + j)
                    + ((lane >= 16) ? 8 : 0);
    int n = nt * 16 + (lane & 15);
    float am = absmax[0];
    float inv = (am > 0.f) ? (127.f / am) : 0.f;
    signed char q = 0;
    if (k < K_REAL)
        q = (signed char)quant8(w[n * K_REAL + k], inv);
    Bp[idx] = q;
}

__global__ void __launch_bounds__(128)
k_gemm(const signed char* __restrict__ Ap,
       const signed char* __restrict__ Bp,
       const float* __restrict__ bias,
       const float* __restrict__ absmax,
       float* __restrict__ out) {
    int lane  = threadIdx.x & 31;
    int wave  = threadIdx.x >> 5;
    int mtile = blockIdx.x * 4 + wave;
    int row   = mtile * 16 + (lane & 15);
    int koff  = (lane >= 16) ? 8 : 0;
    const signed char* aBase = Ap + row * K_PAD + koff;

    v8i acc[4] = {};

    #pragma unroll
    for (int kt = 0; kt < 5; ++kt) {
        const unsigned long long* ap =
            (const unsigned long long*)(aBase + kt * 64);
        if (kt < 4) __builtin_prefetch(aBase + (kt + 1) * 64, 0, 3);
        unsigned long long a0 = ap[0];
        unsigned long long a1 = ap[2];
        unsigned long long a2 = ap[4];
        unsigned long long a3 = ap[6];
        v8i A;
        A[0] = (int)(unsigned)a0; A[1] = (int)(a0 >> 32);
        A[2] = (int)(unsigned)a1; A[3] = (int)(a1 >> 32);
        A[4] = (int)(unsigned)a2; A[5] = (int)(a2 >> 32);
        A[6] = (int)(unsigned)a3; A[7] = (int)(a3 >> 32);
        #pragma unroll
        for (int nt = 0; nt < 4; ++nt) {
            const v8i* bp =
                (const v8i*)(Bp + (((nt * 5 + kt) * 32 + lane) << 5));
            v8i B = *bp;
            acc[nt] = __builtin_amdgcn_wmma_i32_16x16x64_iu8(
                true, A, true, B, acc[nt], false, false);
        }
    }

    float s = (absmax[0] * (1.f / 127.f)) * (absmax[1] * (1.f / 127.f));
    int mrow0 = mtile * 16 + ((lane >= 16) ? 8 : 0);
    int n0    = lane & 15;
    #pragma unroll
    for (int nt = 0; nt < 4; ++nt) {
        int o = nt * 16 + n0;
        float bo = bias[o];
        #pragma unroll
        for (int i = 0; i < 8; ++i) {
            int m = mrow0 + i;
            int b = m >> 10, pix = m & 1023;
            out[((b << 6) + o) * 1024 + pix] = (float)acc[nt][i] * s + bo;
        }
    }
}

extern "C" void kernel_launch(void* const* d_in, const int* in_sizes, int n_in,
                              void* d_out, int out_size, void* d_ws,
                              size_t ws_size, hipStream_t stream) {
    const float* x    = (const float*)d_in[0];
    const float* w    = (const float*)d_in[1];
    const float* bias = (const float*)d_in[2];
    float* out = (float*)d_out;

    char* ws = (char*)d_ws;
    float*       absmax = (float*)ws;
    signed char* Ap     = (signed char*)(ws + 256);
    signed char* Bp     = (signed char*)(ws + 256 + M_TOT * K_PAD);

    int nx = in_sizes[0];
    int nw = in_sizes[1];

    k_zero  <<<1, 32, 0, stream>>>(absmax);
    k_absmax<<<64, 256, 0, stream>>>(x, nx, (unsigned*)absmax);
    k_absmax<<<16, 256, 0, stream>>>(w, nw, (unsigned*)(absmax + 1));
    k_pack_a<<<(M_TOT * K_PAD) / 256, 256, 0, stream>>>(x, absmax, Ap);
    k_pack_b<<<80, 256, 0, stream>>>(w, absmax + 1, Bp);
    k_gemm  <<<64, 128, 0, stream>>>(Ap, Bp, bias, absmax, out);
}
